// SparseTransformer_24816321036377
// MI455X (gfx1250) — matrix operand order repaired and checked by reading, not compiled, not run on hardware
//
#include <hip/hip_runtime.h>
#include <hip/hip_bf16.h>

typedef __bf16 bf16;
typedef __attribute__((ext_vector_type(16))) __bf16 v16bf;
typedef __attribute__((ext_vector_type(8)))  __bf16 v8bf;
typedef __attribute__((ext_vector_type(8)))  float  v8f;
typedef __attribute__((ext_vector_type(4)))  float  v4f;

constexpr int EPI_F32 = 0;             // store f32 accumulators
constexpr int EPI_BIAS_SCALE_BF16 = 1; // (acc + bias[col]) * scale -> bf16

// ---------------------------------------------------------------------------
// Tiled WMMA GEMM: C[M,N] = A[M,K] @ B^T, with B stored [N,K] bf16 row-major.
// N, K compile-time; M via gridDim.y*128. Block tile 128x128, K-step 32,
// 256 threads = 8 waves (2x4 over MxN); wave = 4x2 v_wmma_f32_16x16x32_bf16.
// Double-buffered LDS; all LDS staging is vectorized (b128), no scatter.
// f32 A sources are converted to bf16 while staging into LDS.
// ---------------------------------------------------------------------------
template<bool A_BF16, int EPI, int N, int K>
__global__ __launch_bounds__(256)
void gemm_wmma(const void* __restrict__ Aptr, const bf16* __restrict__ Bptr,
               void* __restrict__ Cptr, const float* __restrict__ bias,
               float scale)
{
    constexpr int BM = 128, BN = 128, BK = 32, PAD = 8; // 80B rows: 16B aligned
    __shared__ __align__(16) bf16 As[2][BM][BK + PAD];
    __shared__ __align__(16) bf16 Bs[2][BN][BK + PAD];

    const int tid  = threadIdx.x;
    const int lane = tid & 31;
    const int wave = tid >> 5;
    const int wm   = (wave & 1) * 64;
    const int wn   = (wave >> 1) * 32;
    const int m0   = blockIdx.y * BM;
    const int n0   = blockIdx.x * BN;
    const int rsel = lane & 15;
    const int hsel = lane >> 4;

    v8f acc[4][2] = {};

    // ---- per-thread global source bases (computed once) ----
    const int a_row = tid >> 1;
    const int a_col = (tid & 1) * 16;
    const bf16*  Abh = (const bf16*) Aptr + (size_t)(m0 + a_row) * K + a_col;
    const float* Afp = (const float*)Aptr + (size_t)(m0 + a_row) * K + a_col;

    const int b_n = tid >> 1;
    const int b_k = (tid & 1) * 16;
    const bf16* Bb = Bptr + (size_t)(n0 + b_n) * K + b_k;

    // staging registers (unused path eliminated per instantiation)
    v8bf hA0, hA1, hB0, hB1;
    v4f  fA[4];

    auto loadA = [&](int k0) {
        if (A_BF16) {
            const v8bf* s = (const v8bf*)(Abh + k0);
            hA0 = s[0]; hA1 = s[1];
        } else {
            const v4f* s = (const v4f*)(Afp + k0);
#pragma unroll
            for (int q = 0; q < 4; q++) fA[q] = s[q];
        }
    };
    auto storeA = [&](int buf) {
        bf16* d = &As[buf][a_row][a_col];
        if (A_BF16) {
            *(v8bf*)d = hA0; *(v8bf*)(d + 8) = hA1;
        } else {
#pragma unroll
            for (int q = 0; q < 4; q++)
#pragma unroll
                for (int j = 0; j < 4; j++) d[q * 4 + j] = (bf16)fA[q][j];
        }
    };
    auto loadB = [&](int k0) {
        const v8bf* s = (const v8bf*)(Bb + k0);
        hB0 = s[0]; hB1 = s[1];
    };
    auto storeB = [&](int buf) {
        bf16* d = &Bs[buf][b_n][b_k];
        *(v8bf*)d = hB0; *(v8bf*)(d + 8) = hB1;
    };
    auto compute = [&](int buf) {
        v16bf afrag[4], bfrag[2];
#pragma unroll
        for (int mi = 0; mi < 4; mi++) {
            const bf16* p = &As[buf][wm + mi * 16 + rsel][0];
#pragma unroll
            for (int e = 0; e < 8; e++) {
                afrag[mi][e]     = p[hsel * 8 + e];       // K 0-7  / 8-15
                afrag[mi][8 + e] = p[16 + hsel * 8 + e];  // K 16-23 / 24-31
            }
        }
#pragma unroll
        for (int ni = 0; ni < 2; ni++) {
            const bf16* p = &Bs[buf][wn + ni * 16 + rsel][0];
#pragma unroll
            for (int e = 0; e < 8; e++) {
                bfrag[ni][e]     = p[hsel * 8 + e];
                bfrag[ni][8 + e] = p[16 + hsel * 8 + e];
            }
        }
#pragma unroll
        for (int mi = 0; mi < 4; mi++)
#pragma unroll
            for (int ni = 0; ni < 2; ni++)
                acc[mi][ni] = __builtin_amdgcn_wmma_f32_16x16x32_bf16(
                    false, afrag[mi], false, bfrag[ni],
                    (short)0, acc[mi][ni], false, false);
    };

    // ---- double-buffered main loop ----
    loadA(0); loadB(0);
    storeA(0); storeB(0);
    __syncthreads();

    constexpr int NT = K / BK;
    int buf = 0;
    for (int kt = 0; kt < NT; kt++, buf ^= 1) {
        if (kt + 1 < NT) {
            loadA((kt + 1) * BK);
            loadB((kt + 1) * BK);
            if (kt + 2 < NT) { // gfx1250 prefetch, WGP scope (all cache levels)
                __builtin_prefetch(A_BF16 ? (const void*)(Abh + (kt + 2) * BK)
                                          : (const void*)(Afp + (kt + 2) * BK), 0, 3);
                __builtin_prefetch((const void*)(Bb + (kt + 2) * BK), 0, 3);
            }
        }
        compute(buf);
        if (kt + 1 < NT) { storeA(buf ^ 1); storeB(buf ^ 1); }
        __syncthreads();
    }

    // ---- epilogue: one base pointer per lane, all offsets are immediates ----
    if (EPI == EPI_F32) {
        float* c = (float*)Cptr + (size_t)(m0 + wm + 8 * hsel) * N + (n0 + wn + rsel);
#pragma unroll
        for (int mi = 0; mi < 4; mi++)
#pragma unroll
            for (int ni = 0; ni < 2; ni++)
#pragma unroll
                for (int r = 0; r < 8; r++)
                    c[(size_t)(mi * 16 + r) * N + ni * 16] = acc[mi][ni][r];
    } else {
        const float b0 = bias[n0 + wn + rsel];
        const float b1 = bias[n0 + wn + 16 + rsel];
        bf16* c = (bf16*)Cptr + (size_t)(m0 + wm + 8 * hsel) * N + (n0 + wn + rsel);
#pragma unroll
        for (int mi = 0; mi < 4; mi++)
#pragma unroll
            for (int r = 0; r < 8; r++) {
                c[(size_t)(mi * 16 + r) * N]      = (bf16)((acc[mi][0][r] + b0) * scale);
                c[(size_t)(mi * 16 + r) * N + 16] = (bf16)((acc[mi][1][r] + b1) * scale);
            }
    }
}

// ---------------------------------------------------------------------------
// Tiled transpose + f32->bf16 convert: out[C,R] = (bf16) in[R,C]^T.
// 32x32 tiles via LDS, coalesced both sides. R, C multiples of 32.
// ---------------------------------------------------------------------------
__global__ __launch_bounds__(256)
void transpose_cvt(const float* __restrict__ in, bf16* __restrict__ out,
                   int R, int C)
{
    __shared__ float sh[32][33];
    const int r0 = blockIdx.y * 32;
    const int c0 = blockIdx.x * 32;
    const int tr = threadIdx.x >> 3;        // 0..31
    const int tc = (threadIdx.x & 7) * 4;   // 0,4,...,28

    v4f v = *(const v4f*)(in + (size_t)(r0 + tr) * C + c0 + tc);
#pragma unroll
    for (int j = 0; j < 4; j++) sh[tr][tc + j] = v[j];
    __syncthreads();

    bf16* dst = out + (size_t)(c0 + tr) * R + r0 + tc;
#pragma unroll
    for (int j = 0; j < 4; j++) dst[j] = (bf16)sh[tc + j][tr];
}

// ---------------------------------------------------------------------------
// Straight f32 -> bf16 convert, 4 elements/thread.
// ---------------------------------------------------------------------------
__global__ __launch_bounds__(256)
void cvt_kernel(const float* __restrict__ in, bf16* __restrict__ out)
{
    const int i = (blockIdx.x * 256 + threadIdx.x) * 4;
    v4f v = *(const v4f*)(in + i);
#pragma unroll
    for (int j = 0; j < 4; j++) out[i + j] = (bf16)v[j];
}

// ---------------------------------------------------------------------------
// Row LayerNorm (biased var, eps=1e-5). One block per row, C = PER*256.
// Optionally writes bf16 and/or f32 results (out_f MAY alias in).
// ---------------------------------------------------------------------------
template<int PER>
__global__ __launch_bounds__(256)
void ln_kernel(const float* in, bf16* out_b, float* out_f, float scale)
{
    const int C   = PER * 256;
    const int row = blockIdx.x;
    const int tid = threadIdx.x;
    const float* rp = in + (size_t)row * C;

    float v[PER];
    float s = 0.f;
#pragma unroll
    for (int i = 0; i < PER; i++) { v[i] = rp[tid + (i << 8)]; s += v[i]; }

    __shared__ float red[256];
    red[tid] = s; __syncthreads();
    for (int off = 128; off > 0; off >>= 1) {
        if (tid < off) red[tid] += red[tid + off];
        __syncthreads();
    }
    const float mean = red[0] / (float)C;
    __syncthreads();

    float s2 = 0.f;
#pragma unroll
    for (int i = 0; i < PER; i++) { float d = v[i] - mean; s2 += d * d; }
    red[tid] = s2; __syncthreads();
    for (int off = 128; off > 0; off >>= 1) {
        if (tid < off) red[tid] += red[tid + off];
        __syncthreads();
    }
    const float rstd = rsqrtf(red[0] / (float)C + 1e-5f);

#pragma unroll
    for (int i = 0; i < PER; i++) {
        const float o = (v[i] - mean) * rstd * scale;
        const size_t idx = (size_t)row * C + tid + (i << 8);
        if (out_b) out_b[idx] = (bf16)o;
        if (out_f) out_f[idx] = o;
    }
}

// ---------------------------------------------------------------------------
// Row softmax over 16384 columns, in place. 64 values/thread in registers.
// ---------------------------------------------------------------------------
__global__ __launch_bounds__(256)
void softmax_kernel(float* __restrict__ data)
{
    const int NC  = 16384;
    const int row = blockIdx.x;
    const int tid = threadIdx.x;
    float* rp = data + (size_t)row * NC;

    float v[64];
    const v4f* src = (const v4f*)rp;
#pragma unroll
    for (int i = 0; i < 16; i++) {
        v4f t = src[tid + (i << 8)];
#pragma unroll
        for (int j = 0; j < 4; j++) v[i * 4 + j] = t[j];
    }

    float mx = -3.4e38f;
#pragma unroll
    for (int i = 0; i < 64; i++) mx = fmaxf(mx, v[i]);

    __shared__ float red[256];
    red[tid] = mx; __syncthreads();
    for (int off = 128; off > 0; off >>= 1) {
        if (tid < off) red[tid] = fmaxf(red[tid], red[tid + off]);
        __syncthreads();
    }
    mx = red[0]; __syncthreads();

    float s = 0.f;
#pragma unroll
    for (int i = 0; i < 64; i++) { v[i] = __expf(v[i] - mx); s += v[i]; }
    red[tid] = s; __syncthreads();
    for (int off = 128; off > 0; off >>= 1) {
        if (tid < off) red[tid] += red[tid + off];
        __syncthreads();
    }
    const float inv = 1.f / red[0];

    v4f* dst = (v4f*)rp;
#pragma unroll
    for (int i = 0; i < 16; i++) {
        v4f t;
#pragma unroll
        for (int j = 0; j < 4; j++) t[j] = v[i * 4 + j] * inv;
        dst[tid + (i << 8)] = t;
    }
}

// ---------------------------------------------------------------------------
extern "C" void kernel_launch(void* const* d_in, const int* in_sizes, int n_in,
                              void* d_out, int out_size, void* d_ws, size_t ws_size,
                              hipStream_t stream)
{
    const float* x   = (const float*)d_in[0];   // [4096, 768]
    const float* Xc  = (const float*)d_in[1];   // [16384, 768]
    const float* Wq  = (const float*)d_in[2];   // [768, 256]
    const float* Wk  = (const float*)d_in[3];   // [768, 256]
    const float* Wv  = (const float*)d_in[4];   // [768, 768]
    const float* qpw = (const float*)d_in[5];   // [256, 256]
    const float* kpw = (const float*)d_in[6];   // [256, 256]
    const float* bq  = (const float*)d_in[7];   // [256]
    const float* bk  = (const float*)d_in[8];   // [256]

    const int B = 4096, n = 768, m = 16384, a = 256;
    const float inv_sqrt_a = 0.0625f;           // 1/sqrt(256)

    // d_out = concat(x, x_hat, f, V)
    float* out_x    = (float*)d_out;
    float* out_xhat = out_x    + (size_t)B * n;
    float* out_f    = out_xhat + (size_t)B * n;
    float* out_V    = out_f    + (size_t)B * m;

    // workspace (~66 MB)
    char* ws = (char*)d_ws;
    size_t off = 0;
    auto alloc = [&](size_t bytes) -> void* {
        void* p = ws + off;
        off += (bytes + 255) & ~(size_t)255;
        return p;
    };
    float* T1  = (float*)alloc((size_t)B * a * 4);   // x@Wq
    float* T2  = (float*)alloc((size_t)m * a * 4);   // X@Wk
    bf16*  Qb  = (bf16*) alloc((size_t)B * a * 2);   // LN(x@Wq)*a^-1/2
    bf16*  Kb  = (bf16*) alloc((size_t)m * a * 2);   // LN(X@Wk)
    bf16*  Q2b = (bf16*) alloc((size_t)B * a * 2);   // (Q@qpw^T+bq)*a^-1/2
    bf16*  K2b = (bf16*) alloc((size_t)m * a * 2);   // K@kpw^T+bk
    bf16*  WqT = (bf16*) alloc((size_t)a * n * 2);   // Wq^T    [256,768]
    bf16*  WkT = (bf16*) alloc((size_t)a * n * 2);   // Wk^T    [256,768]
    bf16*  WvT = (bf16*) alloc((size_t)n * n * 2);   // Wv^T    [768,768]
    bf16*  qpb = (bf16*) alloc((size_t)a * a * 2);   // qpw bf16 (already [N,K])
    bf16*  kpb = (bf16*) alloc((size_t)a * a * 2);   // kpw bf16
    bf16*  VT  = (bf16*) alloc((size_t)n * m * 2);   // V^T     [768,16384]

    const dim3 blk(256);

    // ---- weight preparation (tiny) ----
    transpose_cvt<<<dim3(a / 32, n / 32), blk, 0, stream>>>(Wq, WqT, n, a);
    transpose_cvt<<<dim3(a / 32, n / 32), blk, 0, stream>>>(Wk, WkT, n, a);
    transpose_cvt<<<dim3(n / 32, n / 32), blk, 0, stream>>>(Wv, WvT, n, n);
    cvt_kernel<<<dim3(a * a / 1024), blk, 0, stream>>>(qpw, qpb);
    cvt_kernel<<<dim3(a * a / 1024), blk, 0, stream>>>(kpw, kpb);

    // 1) T1 = x @ Wq           (= x @ WqT^T)   [4096,256] K=768
    gemm_wmma<false, EPI_F32, 256, 768>
        <<<dim3(2, 32), blk, 0, stream>>>(x, WqT, T1, nullptr, 1.f);
    // 2) Qb = LN(T1) * a^-1/2
    ln_kernel<1><<<dim3(B), blk, 0, stream>>>(T1, Qb, nullptr, inv_sqrt_a);
    // 3) T2 = Xc @ Wk                          [16384,256] K=768
    gemm_wmma<false, EPI_F32, 256, 768>
        <<<dim3(2, 128), blk, 0, stream>>>(Xc, WkT, T2, nullptr, 1.f);
    // 4) Kb = LN(T2)
    ln_kernel<1><<<dim3(m), blk, 0, stream>>>(T2, Kb, nullptr, 1.f);
    // 5) out_V(raw) = Xc @ Wv                  [16384,768] K=768
    gemm_wmma<false, EPI_F32, 768, 768>
        <<<dim3(6, 128), blk, 0, stream>>>(Xc, WvT, out_V, nullptr, 1.f);
    // 6) V = LN(raw) -> d_out (f32, in place)
    ln_kernel<3><<<dim3(m), blk, 0, stream>>>(out_V, nullptr, out_V, 1.f);
    // 7) VT = V^T bf16                          [768,16384]
    transpose_cvt<<<dim3(n / 32, m / 32), blk, 0, stream>>>(out_V, VT, m, n);
    // 8) Q2b = (Qb @ qpw^T + bq) * a^-1/2      [4096,256] K=256
    gemm_wmma<true, EPI_BIAS_SCALE_BF16, 256, 256>
        <<<dim3(2, 32), blk, 0, stream>>>(Qb, qpb, Q2b, bq, inv_sqrt_a);
    // 9) K2b = Kb @ kpw^T + bk                 [16384,256] K=256
    gemm_wmma<true, EPI_BIAS_SCALE_BF16, 256, 256>
        <<<dim3(2, 128), blk, 0, stream>>>(Kb, kpb, K2b, bk, 1.f);
    // 10) scores = Q2b @ K2b^T -> out_f        [4096,16384] K=256
    gemm_wmma<true, EPI_F32, 16384, 256>
        <<<dim3(128, 32), blk, 0, stream>>>(Q2b, K2b, out_f, nullptr, 1.f);
    // 11) f = softmax(scores) in place
    softmax_kernel<<<dim3(B), blk, 0, stream>>>(out_f);
    // 12) x_hat = f @ V        (= f @ VT^T)    [4096,768] K=16384
    gemm_wmma<false, EPI_F32, 768, 16384>
        <<<dim3(6, 32), blk, 0, stream>>>(out_f, VT, out_xhat, nullptr, 1.f);
    // 13) pass-through output x
    hipMemcpyAsync(out_x, x, (size_t)B * n * sizeof(float),
                   hipMemcpyDeviceToDevice, stream);
}
